// ScaleDotProductAttention_51488067945127
// MI455X (gfx1250) — matrix operand order repaired and checked by reading, not compiled, not run on hardware
//
#include <hip/hip_runtime.h>

namespace {
constexpr int SEQ = 2048;
constexpr int HD  = 64;
constexpr int NH  = 16;
constexpr int NB  = 2;
constexpr int BLOCK_M = 64;   // q rows per workgroup (4 waves x 16)
constexpr int KT = 64;        // keys per tile
constexpr int NIT = SEQ / KT;
constexpr int KPITCH = 72;    // bf16 elems per K row in LDS   (64 + 8 pad)
constexpr int VPITCH = 72;    // bf16 elems per Vt row in LDS  (64 + 8 pad)
constexpr int PPITCH = 72;    // bf16 elems per P row in LDS   (64 + 8 pad)
}

typedef __bf16 bf16;
typedef __attribute__((ext_vector_type(16))) __bf16 v16bf;
typedef __attribute__((ext_vector_type(8)))  __bf16 v8bf;
typedef __attribute__((ext_vector_type(2)))  __bf16 v2bf;
typedef __attribute__((ext_vector_type(8)))  float  v8f;

__global__ __launch_bounds__(128)
void fa2_wmma_kernel(const float* __restrict__ Q, const float* __restrict__ K,
                     const float* __restrict__ V, const int* __restrict__ M,
                     float* __restrict__ O)
{
  // double-buffered K/V tiles for single-barrier software pipeline
  __shared__ __align__(16) bf16 ldsK [2 * KT * KPITCH];   // [buf][key][d]
  __shared__ __align__(16) bf16 ldsVt[2 * HD * VPITCH];   // [buf][d][key]
  __shared__ __align__(16) bf16 ldsP [4 * 16 * PPITCH];   // per-wave [q][key]

  const int tid  = threadIdx.x;
  const int wave = tid >> 5;
  const int lane = tid & 31;
  const int l16  = lane & 15;
  const int half = lane >> 4;

  const int h  = blockIdx.y;
  const int b  = blockIdx.z;
  const int bh = b * NH + h;
  const int qw = blockIdx.x * BLOCK_M + wave * 16;

  const float* Qb = Q + (size_t)bh * SEQ * HD;
  const float* Kb = K + (size_t)bh * SEQ * HD;
  const float* Vb = V + (size_t)bh * SEQ * HD;
  const int*   Mb = M + (size_t)b * SEQ * SEQ;
  float*       Ob = O + (size_t)bh * SEQ * HD;

  // fold 1/sqrt(64) and log2(e) into Q so softmax uses exp2
  const float scale = 0.125f * 1.44269504088896340736f;
  const float NEG = -1e30f;

  // ---- Q A-operands in 16-bit WMMA A layout (loaded once) ----
  v16bf a0, a1;
  {
    const float* qp = Qb + (size_t)(qw + l16) * HD;
    #pragma unroll
    for (int j = 0; j < 16; ++j) {
      const int d = j + half * 16;
      a0[j] = (__bf16)(qp[d] * scale);
      a1[j] = (__bf16)(qp[d + 32] * scale);
    }
  }

  // all-ones B operand: rowsum(P) = P x ones via WMMA
  v16bf vones;
  #pragma unroll
  for (int j = 0; j < 16; ++j) vones[j] = (__bf16)1.0f;

  v8f o[4] = {{}, {}, {}, {}};
  v8f lacc = {};
  float mrow[8];
  #pragma unroll
  for (int r = 0; r < 8; ++r) mrow[r] = NEG;

  // staging index maps
  const int krow = tid >> 1;          // K: 0..63, two threads per key row
  const int kdch = (tid & 1) * 32;    // K: 32-wide d chunk
  const int vkp  = (tid & 31) * 2;    // V: adjacent key pair base
  const int vdch = (tid >> 5) * 16;   // V: 16-wide d chunk

  // ---- prologue: fetch tile 0 into registers ----
  float4 kreg[8], vreg[8];
  {
    const float* kp  = Kb + (size_t)krow * HD + kdch;
    const float* vp0 = Vb + (size_t)vkp * HD + vdch;
    const float* vp1 = vp0 + HD;
    #pragma unroll
    for (int i = 0; i < 8; ++i) kreg[i] = ((const float4*)kp)[i];
    #pragma unroll
    for (int i = 0; i < 4; ++i) { vreg[i] = ((const float4*)vp0)[i]; vreg[4+i] = ((const float4*)vp1)[i]; }
  }

  for (int it = 0; it < NIT; ++it) {
    const int k0 = it * KT;
    bf16* lk = &ldsK [(it & 1) * (KT * KPITCH)];
    bf16* lv = &ldsVt[(it & 1) * (HD * VPITCH)];

    // ---- drain staged registers into LDS (convert to bf16) ----
    {
      #pragma unroll
      for (int c = 0; c < 4; ++c) {
        const float* kf = (const float*)&kreg[2*c];
        v8bf kv;
        #pragma unroll
        for (int i = 0; i < 8; ++i) kv[i] = (__bf16)kf[i];
        *(v8bf*)&lk[krow * KPITCH + kdch + 8*c] = kv;
      }
      const float* vf0 = (const float*)&vreg[0];
      const float* vf1 = (const float*)&vreg[4];
      #pragma unroll
      for (int i = 0; i < 16; ++i) {
        v2bf pr2;
        pr2[0] = (__bf16)vf0[i];
        pr2[1] = (__bf16)vf1[i];
        *(v2bf*)&lv[(vdch + i) * VPITCH + vkp] = pr2;  // b32 store, key pair
      }
    }

    // ---- issue next tile's global loads (consumed next iteration) ----
    if (it + 1 < NIT) {
      const int kn = k0 + KT;
      const float* kp  = Kb + (size_t)(kn + krow) * HD + kdch;
      const float* vp0 = Vb + (size_t)(kn + vkp) * HD + vdch;
      const float* vp1 = vp0 + HD;
      #pragma unroll
      for (int i = 0; i < 8; ++i) kreg[i] = ((const float4*)kp)[i];
      #pragma unroll
      for (int i = 0; i < 4; ++i) { vreg[i] = ((const float4*)vp0)[i]; vreg[4+i] = ((const float4*)vp1)[i]; }
      if (it + 2 < NIT) {                 // warm L2 for tile i+2
        __builtin_prefetch(kp + (size_t)KT * HD, 0, 3);
        __builtin_prefetch(vp0 + (size_t)KT * HD, 0, 3);
      }
    }

    __syncthreads();   // single barrier: buf (it&1) fully staged, other buf free

    // ---- S = Q·K^T for 64 keys (4 N-subtiles x 2 K-halves) ----
    v8f c[4] = {{}, {}, {}, {}};
    #pragma unroll
    for (int sub = 0; sub < 4; ++sub) {
      const v16bf kb0 = *(const v16bf*)&lk[(16*sub + l16) * KPITCH + half * 16];
      const v16bf kb1 = *(const v16bf*)&lk[(16*sub + l16) * KPITCH + half * 16 + 32];
      c[sub] = __builtin_amdgcn_wmma_f32_16x16x32_bf16(false, a0, false, kb0, (short)0, c[sub], false, false);
      c[sub] = __builtin_amdgcn_wmma_f32_16x16x32_bf16(false, a1, false, kb1, (short)0, c[sub], false, false);
    }

    // ---- masked online softmax (f32, per row r+8*half) ----
    #pragma unroll
    for (int r = 0; r < 8; ++r) {
      const int qr = qw + r + half * 8;
      const int* mp = Mb + (size_t)qr * SEQ + k0 + l16;
      float s[4];
      #pragma unroll
      for (int sub = 0; sub < 4; ++sub)
        s[sub] = mp[16*sub] ? c[sub][r] : NEG;
      float mx = fmaxf(fmaxf(s[0], s[1]), fmaxf(s[2], s[3]));
      mx = fmaxf(mx, __shfl_xor(mx, 1, 32));
      mx = fmaxf(mx, __shfl_xor(mx, 2, 32));
      mx = fmaxf(mx, __shfl_xor(mx, 4, 32));
      mx = fmaxf(mx, __shfl_xor(mx, 8, 32));
      const float mn = fmaxf(mrow[r], mx);
      const float al = exp2f(mrow[r] - mn);
      mrow[r] = mn;
      o[0][r] *= al; o[1][r] *= al; o[2][r] *= al; o[3][r] *= al;
      lacc[r] *= al;
      bf16* pw = &ldsP[(wave * 16 + r + half * 8) * PPITCH];
      #pragma unroll
      for (int sub = 0; sub < 4; ++sub)
        pw[l16 + 16*sub] = (__bf16)exp2f(s[sub] - mn);
    }

    // ---- O += P·V (4 d-tiles x 2 key-groups) + rowsum via ones-WMMA ----
    {
      const bf16* pp = &ldsP[(wave * 16 + l16) * PPITCH];
      union { v16bf v; v8bf hh[2]; } u;
      u.hh[0] = *(const v8bf*)&pp[half * 16];
      u.hh[1] = *(const v8bf*)&pp[half * 16 + 8];
      const v16bf pa0 = u.v;
      u.hh[0] = *(const v8bf*)&pp[32 + half * 16];
      u.hh[1] = *(const v8bf*)&pp[40 + half * 16];
      const v16bf pa1 = u.v;
      #pragma unroll
      for (int t = 0; t < 4; ++t) {
        const v16bf vb0 = *(const v16bf*)&lv[(t*16 + l16) * VPITCH + half * 16];
        const v16bf vb1 = *(const v16bf*)&lv[(t*16 + l16) * VPITCH + 32 + half * 16];
        o[t] = __builtin_amdgcn_wmma_f32_16x16x32_bf16(false, pa0, false, vb0, (short)0, o[t], false, false);
        o[t] = __builtin_amdgcn_wmma_f32_16x16x32_bf16(false, pa1, false, vb1, (short)0, o[t], false, false);
      }
      lacc = __builtin_amdgcn_wmma_f32_16x16x32_bf16(false, pa0, false, vones, (short)0, lacc, false, false);
      lacc = __builtin_amdgcn_wmma_f32_16x16x32_bf16(false, pa1, false, vones, (short)0, lacc, false, false);
    }
    // no trailing barrier: next iteration writes the OTHER buffer; the
    // single barrier per iteration orders buffer reuse two iterations apart
  }

  // ---- epilogue: normalize and store ----
  #pragma unroll
  for (int r = 0; r < 8; ++r) {
    const int qr = qw + r + half * 8;
    const float inv = 1.f / lacc[r];
    float* op = Ob + (size_t)qr * HD + l16;
    op[0]  = o[0][r] * inv;
    op[16] = o[1][r] * inv;
    op[32] = o[2][r] * inv;
    op[48] = o[3][r] * inv;
  }
}

extern "C" void kernel_launch(void* const* d_in, const int* in_sizes, int n_in,
                              void* d_out, int out_size, void* d_ws, size_t ws_size,
                              hipStream_t stream)
{
  (void)in_sizes; (void)n_in; (void)out_size; (void)d_ws; (void)ws_size;
  const float* q    = (const float*)d_in[0];
  const float* k    = (const float*)d_in[1];
  const float* v    = (const float*)d_in[2];
  const int*   mask = (const int*)d_in[3];
  float* out = (float*)d_out;
  dim3 grid(SEQ / BLOCK_M, NH, NB);
  fa2_wmma_kernel<<<grid, 128, 0, stream>>>(q, k, v, mask, out);
}
